// WindowAttention_63548336111768
// MI455X (gfx1250) — matrix operand order repaired and checked by reading, not compiled, not run on hardware
//
#include <hip/hip_runtime.h>

typedef __attribute__((ext_vector_type(16))) _Float16 v16h;
typedef __attribute__((ext_vector_type(8)))  _Float16 v8h;
typedef __attribute__((ext_vector_type(4)))  _Float16 v4h;
typedef __attribute__((ext_vector_type(8)))  float    v8f;

#define DIM   256
#define HEADS 8
#define HD    32
#define NTOK  64

// workspace layout (bytes):
//   [0,       393216): WqkvT  f16 [768][256]   (n-major, i.e. transposed w_qkv)
//   [393216,  524288): WprojT f16 [256][256]   (n-major)
//   [524288,  655360): biasT  f32 [8][64][64]  (bias[h][query][key])
#define WS_WPROJ 393216
#define WS_BIAS  524288

__device__ __forceinline__ v16h cat16(const _Float16* p0, const _Float16* p1) {
  v8h a = *(const v8h*)p0;
  v8h b = *(const v8h*)(p1);
  return __builtin_shufflevector(a, b, 0,1,2,3,4,5,6,7,8,9,10,11,12,13,14,15);
}

// A-matrix tile 16x32 (MxK), src row-major [m][k], stride rs halves.
// Lane holds row m=lane&15; halves 0..7 = K[k0+hi*8..+8), 8..15 = K[k0+16+hi*8..+8).
__device__ __forceinline__ v16h ldA(const _Float16* src, int rs, int m0, int k0,
                                    int lo, int hi) {
  const _Float16* p = src + (m0 + lo) * rs + k0 + hi * 8;
  return cat16(p, p + 16);
}

// B-matrix tile 32x16 (KxN), src stored n-major: src[n][k], stride rs halves.
// Lane holds column n=lane&15; halves 0..7 = K[k0+hi*8..+8), 8..15 = K[k0+16+hi*8..+8).
__device__ __forceinline__ v16h ldB(const _Float16* src, int rs, int n0, int k0,
                                    int lo, int hi) {
  const _Float16* p = src + (n0 + lo) * rs + k0 + hi * 8;
  return cat16(p, p + 16);
}

__global__ void prep_kernel(const float* __restrict__ w_qkv,
                            const float* __restrict__ w_proj,
                            const float* __restrict__ rel_pos_bias,
                            const int*   __restrict__ rel_idx,
                            _Float16* __restrict__ wqkvT,
                            _Float16* __restrict__ wprojT,
                            float*    __restrict__ biasT) {
  int t  = blockIdx.x * blockDim.x + threadIdx.x;
  int nT = gridDim.x * blockDim.x;
  for (int i = t; i < 768 * 256; i += nT) {
    int n = i >> 8, k = i & 255;
    wqkvT[i] = (_Float16)w_qkv[k * 768 + n];
  }
  for (int i = t; i < 256 * 256; i += nT) {
    int n = i >> 8, k = i & 255;
    wprojT[i] = (_Float16)w_proj[k * 256 + n];
  }
  for (int i = t; i < 8 * 64 * 64; i += nT) {
    int h = i >> 12, mn = i & 4095;
    biasT[i] = rel_pos_bias[rel_idx[mn] * 8 + h];
  }
}

__global__ __launch_bounds__(256) void win_attn_kernel(
    const float* __restrict__ x,
    const float* __restrict__ b_qkv,
    const float* __restrict__ b_proj,
    const _Float16* __restrict__ wqkvT,
    const _Float16* __restrict__ wprojT,
    const float* __restrict__ biasT,
    float* __restrict__ out) {
  // Region A (33792 B): X f16 [64][264]  ->  per-wave P stage [8][16*72]  ->  O f16 [64][264]
  __shared__ __align__(32) _Float16 smemA[NTOK * 264];
  // Per head: Q [64][40], K [64][40], Vt [32][72]  (7424 halves each)
  __shared__ __align__(32) _Float16 smemQKV[HEADS * 7424];

  const int tid  = threadIdx.x;
  const int wave = tid >> 5;
  const int lane = tid & 31;
  const int lo   = lane & 15;
  const int hi   = lane >> 4;
  const int win  = blockIdx.x;
  const int h    = wave;

  // ---- Phase 1: load x tile fp32 -> f16 LDS (coalesced float4, packed b64 stores) ----
  const float4* xs = (const float4*)(x + (size_t)win * (NTOK * DIM));
#pragma unroll
  for (int i = 0; i < 16; ++i) {
    int idx4 = i * 256 + tid;
    float4 v = xs[idx4];
    int flat = idx4 * 4;
    int row = flat >> 8, col = flat & 255;
    v4h p;
    p[0] = (_Float16)v.x; p[1] = (_Float16)v.y;
    p[2] = (_Float16)v.z; p[3] = (_Float16)v.w;
    *(v4h*)&smemA[row * 264 + col] = p;
  }
  __syncthreads();

  _Float16* Qs  = &smemQKV[h * 7424];
  _Float16* Ks  = Qs + 64 * 40;
  _Float16* Vts = Ks + 64 * 40;   // transposed: [hd 32][tok 64], stride 72

  // ---- Phase 2a: Q,K via D = W x X^T  (lane's 8 results = 8 consecutive cols
  //      at one token -> packed b128 store into token-major [tok][hd]) ----
#pragma unroll
  for (int j = 0; j < 4; ++j) {
    int n0 = (j < 2 ? 0 : 256) + h * 32 + (j & 1) * 16;
    float bq[8];
#pragma unroll
    for (int r = 0; r < 8; ++r) bq[r] = b_qkv[n0 + 8 * hi + r];
    v16h aw[8];
#pragma unroll
    for (int ks = 0; ks < 8; ++ks) aw[ks] = ldA(wqkvT, 256, n0, ks * 32, lo, hi);
    _Float16* dstBase = (j < 2 ? Qs : Ks);
#pragma unroll
    for (int tt = 0; tt < 4; ++tt) {
      v8f acc = {0, 0, 0, 0, 0, 0, 0, 0};
#pragma unroll
      for (int ks = 0; ks < 8; ++ks) {
        v16h b = ldB(smemA, 264, tt * 16, ks * 32, lo, hi);
        acc = __builtin_amdgcn_wmma_f32_16x16x32_f16(false, aw[ks], false, b,
                                                     (short)0, acc, false, false);
      }
      v8h pk;
#pragma unroll
      for (int r = 0; r < 8; ++r) pk[r] = (_Float16)(acc[r] + bq[r]);
      *(v8h*)&dstBase[(tt * 16 + lo) * 40 + (j & 1) * 16 + 8 * hi] = pk;
    }
  }

  // ---- Phase 2b: V via D = X x W  (lane's 8 results = 8 consecutive tokens at
  //      one head-dim -> packed b128 store into Vt[d][tok]) ----
#pragma unroll
  for (int tm = 0; tm < 4; ++tm) {
    v16h ax[8];
#pragma unroll
    for (int ks = 0; ks < 8; ++ks) ax[ks] = ldA(smemA, 264, tm * 16, ks * 32, lo, hi);
#pragma unroll
    for (int jv = 0; jv < 2; ++jv) {
      int n0 = 512 + h * 32 + jv * 16;
      float bq = b_qkv[n0 + lo];
      v8f acc = {0, 0, 0, 0, 0, 0, 0, 0};
#pragma unroll
      for (int ks = 0; ks < 8; ++ks) {
        v16h b = ldB(wqkvT, 256, n0, ks * 32, lo, hi);
        acc = __builtin_amdgcn_wmma_f32_16x16x32_f16(false, ax[ks], false, b,
                                                     (short)0, acc, false, false);
      }
      v8h pk;
#pragma unroll
      for (int r = 0; r < 8; ++r) pk[r] = (_Float16)(acc[r] + bq);
      *(v8h*)&Vts[(jv * 16 + lo) * 72 + tm * 16 + 8 * hi] = pk;
    }
  }
  __syncthreads();   // QKV done; region A's X no longer needed

  // ---- Phase 3: attention for head h ----
  _Float16* stage = &smemA[wave * (16 * 72)];
  const float scale = 0.17677669529663687f;        // 32^-0.5
  const float* biasH = biasT + h * (64 * 64);

  v16h qa[4], kb[4], va[2][2];
#pragma unroll
  for (int tm = 0; tm < 4; ++tm) qa[tm] = ldA(Qs, 40, tm * 16, 0, lo, hi);
#pragma unroll
  for (int tn = 0; tn < 4; ++tn) kb[tn] = ldB(Ks, 40, tn * 16, 0, lo, hi);
#pragma unroll
  for (int dt = 0; dt < 2; ++dt)
#pragma unroll
    for (int ks = 0; ks < 2; ++ks)
      va[dt][ks] = ldA(Vts, 72, dt * 16, ks * 32, lo, hi);   // A = V^T rows (hd-major)

  v8f S[4][4];
#pragma unroll
  for (int tm = 0; tm < 4; ++tm)
#pragma unroll
    for (int tn = 0; tn < 4; ++tn) {
      v8f z = {0, 0, 0, 0, 0, 0, 0, 0};
      S[tm][tn] = __builtin_amdgcn_wmma_f32_16x16x32_f16(false, qa[tm], false, kb[tn],
                                                         (short)0, z, false, false);
    }

  // scale + bias + row softmax (rows live across the 16 lanes of each half-wave)
#pragma unroll
  for (int tm = 0; tm < 4; ++tm) {
#pragma unroll
    for (int r = 0; r < 8; ++r) {
      int m = tm * 16 + r + 8 * hi;
      float mx = -3.0e38f;
#pragma unroll
      for (int tn = 0; tn < 4; ++tn) {
        float t = S[tm][tn][r] * scale + biasH[m * 64 + tn * 16 + lo];
        S[tm][tn][r] = t;
        mx = fmaxf(mx, t);
      }
#pragma unroll
      for (int off = 1; off < 16; off <<= 1) mx = fmaxf(mx, __shfl_xor(mx, off, 32));
      float sum = 0.0f;
#pragma unroll
      for (int tn = 0; tn < 4; ++tn) {
        float e = __expf(S[tm][tn][r] - mx);
        S[tm][tn][r] = e;
        sum += e;
      }
#pragma unroll
      for (int off = 1; off < 16; off <<= 1) sum += __shfl_xor(sum, off, 32);
      float inv = 1.0f / sum;
#pragma unroll
      for (int tn = 0; tn < 4; ++tn) S[tm][tn][r] *= inv;
    }
  }

  // O^T = V^T x P^T : A = Vt tiles, B = P stage (token-major rows).
  // D-layout of O^T gives 8 consecutive head-dims per lane -> packed writeback.
  v8f OT[4][2];
#pragma unroll
  for (int tm = 0; tm < 4; ++tm)
#pragma unroll
    for (int dt = 0; dt < 2; ++dt) OT[tm][dt] = (v8f){0, 0, 0, 0, 0, 0, 0, 0};

#pragma unroll
  for (int tm = 0; tm < 4; ++tm) {
#pragma unroll
    for (int tn = 0; tn < 4; ++tn)
#pragma unroll
      for (int r = 0; r < 8; ++r)
        stage[(r + 8 * hi) * 72 + tn * 16 + lo] = (_Float16)S[tm][tn][r];
    // DS ops are in-order within a wave: loads below see the stores above
    v16h pb0 = ldB(stage, 72, 0, 0, lo, hi);
    v16h pb1 = ldB(stage, 72, 0, 32, lo, hi);
#pragma unroll
    for (int dt = 0; dt < 2; ++dt) {
      OT[tm][dt] = __builtin_amdgcn_wmma_f32_16x16x32_f16(false, va[dt][0], false, pb0,
                                                          (short)0, OT[tm][dt], false, false);
      OT[tm][dt] = __builtin_amdgcn_wmma_f32_16x16x32_f16(false, va[dt][1], false, pb1,
                                                          (short)0, OT[tm][dt], false, false);
    }
  }
  __syncthreads();   // all waves done with stages; region A becomes O

  // O^T element (d = dt*16 + r + 8*hi, m = tm*16 + lo): 8 consecutive d per lane.
#pragma unroll
  for (int tm = 0; tm < 4; ++tm)
#pragma unroll
    for (int dt = 0; dt < 2; ++dt) {
      v8h pk;
#pragma unroll
      for (int r = 0; r < 8; ++r) pk[r] = (_Float16)OT[tm][dt][r];
      *(v8h*)&smemA[(tm * 16 + lo) * 264 + h * 32 + dt * 16 + 8 * hi] = pk;
    }
  __syncthreads();

  // ---- Phase 4: output projection; wave w owns output columns [w*32, w*32+32) ----
  float* op = out + (size_t)win * (NTOK * DIM);
  const int nbase = wave * 32;
#pragma unroll
  for (int tm = 0; tm < 4; ++tm) {
    v16h a[8];
#pragma unroll
    for (int ks = 0; ks < 8; ++ks) a[ks] = ldA(smemA, 264, tm * 16, ks * 32, lo, hi);
#pragma unroll
    for (int jn = 0; jn < 2; ++jn) {
      int n0 = nbase + jn * 16;
      float bp = b_proj[n0 + lo];
      v8f acc = {0, 0, 0, 0, 0, 0, 0, 0};
#pragma unroll
      for (int ks = 0; ks < 8; ++ks) {
        v16h b = ldB(wprojT, 256, n0, ks * 32, lo, hi);
        acc = __builtin_amdgcn_wmma_f32_16x16x32_f16(false, a[ks], false, b,
                                                     (short)0, acc, false, false);
      }
#pragma unroll
      for (int r = 0; r < 8; ++r)
        op[(tm * 16 + r + 8 * hi) * 256 + n0 + lo] = acc[r] + bp;
    }
  }
}

extern "C" void kernel_launch(void* const* d_in, const int* in_sizes, int n_in,
                              void* d_out, int out_size, void* d_ws, size_t ws_size,
                              hipStream_t stream) {
  const float* x            = (const float*)d_in[0];
  const float* w_qkv        = (const float*)d_in[1];
  const float* b_qkv        = (const float*)d_in[2];
  const float* w_proj       = (const float*)d_in[3];
  const float* b_proj       = (const float*)d_in[4];
  const float* rel_pos_bias = (const float*)d_in[5];
  const int*   rel_idx      = (const int*)d_in[6];

  _Float16* wqkvT  = (_Float16*)d_ws;
  _Float16* wprojT = (_Float16*)((char*)d_ws + WS_WPROJ);
  float*    biasT  = (float*)((char*)d_ws + WS_BIAS);

  int nwin = in_sizes[0] / (NTOK * DIM);

  prep_kernel<<<256, 256, 0, stream>>>(w_qkv, w_proj, rel_pos_bias, rel_idx,
                                       wqkvT, wprojT, biasT);
  win_attn_kernel<<<nwin, 256, 0, stream>>>(x, b_qkv, b_proj, wqkvT, wprojT,
                                            biasT, (float*)d_out);
}
